// EdgeMidpointEncoder_18451179504417
// MI455X (gfx1250) — matrix operand order repaired and checked by reading, not compiled, not run on hardware
//
#include <hip/hip_runtime.h>
#include <hip/hip_bf16.h>

typedef __attribute__((ext_vector_type(16))) _Float16 v16h;
typedef __attribute__((ext_vector_type(8)))  _Float16 v8h;
typedef __attribute__((ext_vector_type(4)))  _Float16 v4h;
typedef __attribute__((ext_vector_type(8)))  float    v8f;

union AFrag { v16h v; v8h h[2]; };

// Opaque |x|: keeps the SLP vectorizer from fusing sub+abs into the
// v2f32 and-mask pattern that crashes gfx1250 ISel (clang-23).
__device__ __forceinline__ float opaque_fabs(float x) {
    float r;
    __asm__("v_and_b32 %0, 0x7fffffff, %1" : "=v"(r) : "v"(x));
    return r;
}

// ---- packed weight image layout (bytes) in d_ws and in LDS ----
#define W1FRAG_OFF 0           // 6 chunks * 16 tiles * 1024B (f16 B-fragments, [W1s|W1v])
#define W192_OFF   98304       // 256 f32: row K=192 of [W1s|W1v] (edge-length column)
#define W2FRAG_OFF 99328       // 4 chunks * 6 tiles * 1024B   ([W2s(4)|W2v(2)])
#define B1_OFF     123904      // 256 f32: [b1s|b1v]
#define B2S_OFF    124928      // 64 f32
#define B2V_OFF    125184      // 32 f32
#define WS_BYTES   125312

// ---- per-wave LDS scratch (M=32 edge tile per wave) ----
#define IN_OFF     0           // 32 rows x 192 f16 = 12288B (input tile)
#define H_OFF      12288       // 32 rows x 256 f16 = 16384B (hidden tile, scalar|vector)
#define LEN_OFF    28672       // 32 f32 edge lengths
#define CS_OFF     28800       // 32 float2 (cos,sin)
#define PWAVE      29056
#define NWAVES     4
#define TILE_M     32
#define LDS_TOTAL  (WS_BYTES + NWAVES * PWAVE)   // 241536 bytes

// =====================================================================
// Weight repack: f32 row-major -> f16 WMMA B-fragments (+ biases, K=192 row)
// B-fragment (32x16, f16): lane = n + 16*((k>>3)&1), halfIdx = (k&7) + 8*(k>>4)  (ISA 7.12.2)
// =====================================================================
__global__ void prep_weights_kernel(const float* __restrict__ W1s,
                                    const float* __restrict__ b1s,
                                    const float* __restrict__ W2s,
                                    const float* __restrict__ b2s,
                                    const float* __restrict__ W1v,
                                    const float* __restrict__ b1v,
                                    const float* __restrict__ W2v,
                                    const float* __restrict__ b2v,
                                    char* __restrict__ ws)
{
    _Float16* wh = (_Float16*)ws;
    float*    wf = (float*)ws;
    const int TOT_A = 6 * 16 * 32 * 16;   // 49152 W1 fragment elements
    const int TOT_C = 4 * 6 * 32 * 16;    // 12288 W2 fragment elements
    const int TOT   = TOT_A + TOT_C + 256 + 256 + 64 + 32;
    for (int idx = blockIdx.x * blockDim.x + threadIdx.x; idx < TOT;
         idx += gridDim.x * blockDim.x) {
        if (idx < TOT_A) {
            int e = idx & 15, lane = (idx >> 4) & 31, t = (idx >> 9) & 15, c = idx >> 13;
            int k  = c * 32 + (e >> 3) * 16 + (lane >> 4) * 8 + (e & 7);     // 0..191
            int nl = lane & 15;
            float v = (t < 8) ? W1s[k * 128 + t * 16 + nl]
                              : W1v[k * 128 + (t - 8) * 16 + nl];
            wh[(c * 16 + t) * 512 + lane * 16 + e] = (_Float16)v;
        } else if (idx < TOT_A + TOT_C) {
            int i = idx - TOT_A;
            int e = i & 15, lane = (i >> 4) & 31;
            int rest = i >> 9;                          // 0..23
            int t = rest % 6, c = rest / 6;
            int k  = c * 32 + (e >> 3) * 16 + (lane >> 4) * 8 + (e & 7);     // 0..127
            int nl = lane & 15;
            float v = (t < 4) ? W2s[k * 64 + t * 16 + nl]
                              : W2v[k * 32 + (t - 4) * 16 + nl];
            wh[W2FRAG_OFF / 2 + (c * 6 + t) * 512 + lane * 16 + e] = (_Float16)v;
        } else if (idx < TOT_A + TOT_C + 256) {
            int n = idx - (TOT_A + TOT_C);
            wf[W192_OFF / 4 + n] = (n < 128) ? W1s[192 * 128 + n]
                                             : W1v[192 * 128 + (n - 128)];
        } else if (idx < TOT_A + TOT_C + 512) {
            int n = idx - (TOT_A + TOT_C + 256);
            wf[B1_OFF / 4 + n] = (n < 128) ? b1s[n] : b1v[n - 128];
        } else if (idx < TOT_A + TOT_C + 512 + 64) {
            int n = idx - (TOT_A + TOT_C + 512);
            wf[B2S_OFF / 4 + n] = b2s[n];
        } else {
            int n = idx - (TOT_A + TOT_C + 512 + 64);
            wf[B2V_OFF / 4 + n] = b2v[n];
        }
    }
}

// One 4-N-tile x 2-M-block pass of layer 1 (64 accumulator VGPRs live).
// Each B-fragment is loaded once and feeds TWO WMMAs (rows 0-15 and 16-31),
// halving LDS weight traffic per edge.
__device__ __forceinline__
void layer1_pass(const _Float16* __restrict__ inT,
                 const _Float16* __restrict__ w1f,
                 const float* __restrict__ w192,
                 const float* __restrict__ b1,
                 const float* __restrict__ lnL,
                 _Float16* __restrict__ hT,
                 int nl, int hi, int lane, int tbase)
{
    const v8f zero = {0.f, 0.f, 0.f, 0.f, 0.f, 0.f, 0.f, 0.f};
    v8f acc[4][2];
#pragma unroll
    for (int t = 0; t < 4; ++t) { acc[t][0] = zero; acc[t][1] = zero; }

#pragma unroll
    for (int c = 0; c < 6; ++c) {
        AFrag A0, A1;
        const _Float16* ar0 = inT + nl * 192 + c * 32 + hi * 8;
        const _Float16* ar1 = ar0 + 16 * 192;
        A0.h[0] = *(const v8h*)ar0;  A0.h[1] = *(const v8h*)(ar0 + 16);
        A1.h[0] = *(const v8h*)ar1;  A1.h[1] = *(const v8h*)(ar1 + 16);
#pragma unroll
        for (int t = 0; t < 4; ++t) {
            AFrag B;
            const _Float16* br = w1f + (c * 16 + tbase + t) * 512 + lane * 16;
            B.h[0] = *(const v8h*)br;
            B.h[1] = *(const v8h*)(br + 8);
            acc[t][0] = __builtin_amdgcn_wmma_f32_16x16x32_f16(
                false, A0.v, false, B.v, (short)0, acc[t][0], false, false);
            acc[t][1] = __builtin_amdgcn_wmma_f32_16x16x32_f16(
                false, A1.v, false, B.v, (short)0, acc[t][1], false, false);
        }
    }

    // rank-1 (K=192 length column) + bias + ReLU -> h tile slice (f16)
#pragma unroll
    for (int t = 0; t < 4; ++t) {
        int tg = tbase + t;
        float w  = w192[tg * 16 + nl];
        float bb = b1[tg * 16 + nl];
#pragma unroll
        for (int mb = 0; mb < 2; ++mb) {
#pragma unroll
            for (int r = 0; r < 8; ++r) {
                int m = r + 8 * hi + 16 * mb;
                float v = acc[t][mb][r] + lnL[m] * w + bb;
                v = v > 0.f ? v : 0.f;
                hT[m * 256 + tg * 16 + nl] = (_Float16)v;
            }
        }
    }
}

// =====================================================================
// Main kernel: each wave processes a 32-edge tile through both MLPs
// via v_wmma_f32_16x16x32_f16 with LDS-resident weights.
// =====================================================================
__global__ __launch_bounds__(NWAVES * 32, 1)
void edge_mlp_kernel(const long long* __restrict__ ep,     // [E,2] int64
                     const float* __restrict__ lens,       // [E,1]
                     const float* __restrict__ theta,      // [E]
                     const float* __restrict__ feat,       // [N,64]
                     const char* __restrict__ wsImg,       // packed weights
                     float* __restrict__ out,              // h0 | v0
                     int nEdges, int nTiles)
{
    extern __shared__ char smem[];

    // --- stage weight image into LDS (whole block) ---
    {
        float4* dst = (float4*)smem;
        const float4* src = (const float4*)wsImg;
        for (int i = threadIdx.x; i < WS_BYTES / 16; i += blockDim.x)
            dst[i] = src[i];
    }
    __syncthreads();

    const int wave = threadIdx.x >> 5;
    const int lane = threadIdx.x & 31;
    const int nl   = lane & 15;
    const int hi   = lane >> 4;

    char* wbase = smem + WS_BYTES + wave * PWAVE;
    _Float16* inT = (_Float16*)(wbase + IN_OFF);
    _Float16* hT  = (_Float16*)(wbase + H_OFF);
    float*    lnL = (float*)(wbase + LEN_OFF);
    float2*   csL = (float2*)(wbase + CS_OFF);

    const _Float16* w1f  = (const _Float16*)(smem + W1FRAG_OFF);
    const float*    w192 = (const float*)(smem + W192_OFF);
    const _Float16* w2f  = (const _Float16*)(smem + W2FRAG_OFF);
    const float*    b1   = (const float*)(smem + B1_OFF);
    const float*    b2s  = (const float*)(smem + B2S_OFF);
    const float*    b2v  = (const float*)(smem + B2V_OFF);

    const size_t V0_BASE = (size_t)nEdges * 64;
    const v8f zero = {0.f, 0.f, 0.f, 0.f, 0.f, 0.f, 0.f, 0.f};

    for (int tile = blockIdx.x * NWAVES + wave; tile < nTiles;
         tile += gridDim.x * NWAVES) {
        const int e0 = tile * TILE_M;

        // ---- gather: 1 lane per edge row; packed v4h (8B) LDS stores ----
        {
            int m = lane;                       // 0..31
            int e = min(e0 + m, nEdges - 1);
            long long vi = ep[2 * (long long)e + 0];
            long long vj = ep[2 * (long long)e + 1];
            const float4* fi = (const float4*)(feat + vi * 64);
            const float4* fj = (const float4*)(feat + vj * 64);
            _Float16* row = inT + m * 192;
#pragma unroll
            for (int q = 0; q < 16; ++q) {
                float4 a = fi[q], b = fj[q];
                int o = q * 4;
                v4h pa = {(_Float16)a.x, (_Float16)a.y, (_Float16)a.z, (_Float16)a.w};
                v4h pb = {(_Float16)b.x, (_Float16)b.y, (_Float16)b.z, (_Float16)b.w};
                float d0 = opaque_fabs(a.x - b.x);
                float d1 = opaque_fabs(a.y - b.y);
                float d2 = opaque_fabs(a.z - b.z);
                float d3 = opaque_fabs(a.w - b.w);
                v4h pd = {(_Float16)d0, (_Float16)d1, (_Float16)d2, (_Float16)d3};
                *(v4h*)(row + o)       = pa;
                *(v4h*)(row + 64 + o)  = pb;
                *(v4h*)(row + 128 + o) = pd;
            }
            lnL[m] = lens[e];
            float s, c;
            __sincosf(theta[e], &s, &c);
            csL[m] = make_float2(c, s);
        }
        __asm__ volatile("s_wait_dscnt 0" ::: "memory");   // same-wave LDS RAW

        // ---- layer 1 in four 4-N-tile passes (keeps VGPR pressure low) ----
        layer1_pass(inT, w1f, w192, b1, lnL, hT, nl, hi, lane, 0);
        layer1_pass(inT, w1f, w192, b1, lnL, hT, nl, hi, lane, 4);
        layer1_pass(inT, w1f, w192, b1, lnL, hT, nl, hi, lane, 8);
        layer1_pass(inT, w1f, w192, b1, lnL, hT, nl, hi, lane, 12);
        __asm__ volatile("s_wait_dscnt 0" ::: "memory");

        // ---- layer 2, scalar head: h_s(32x128) @ W2s(128x64) ----
        {
            v8f acc2[4][2];
#pragma unroll
            for (int t = 0; t < 4; ++t) { acc2[t][0] = zero; acc2[t][1] = zero; }
#pragma unroll
            for (int c = 0; c < 4; ++c) {
                AFrag A0, A1;
                const _Float16* hr0 = hT + nl * 256 + c * 32 + hi * 8;
                const _Float16* hr1 = hr0 + 16 * 256;
                A0.h[0] = *(const v8h*)hr0;  A0.h[1] = *(const v8h*)(hr0 + 16);
                A1.h[0] = *(const v8h*)hr1;  A1.h[1] = *(const v8h*)(hr1 + 16);
#pragma unroll
                for (int t = 0; t < 4; ++t) {
                    AFrag B;
                    const _Float16* br = w2f + (c * 6 + t) * 512 + lane * 16;
                    B.h[0] = *(const v8h*)br;
                    B.h[1] = *(const v8h*)(br + 8);
                    acc2[t][0] = __builtin_amdgcn_wmma_f32_16x16x32_f16(
                        false, A0.v, false, B.v, (short)0, acc2[t][0], false, false);
                    acc2[t][1] = __builtin_amdgcn_wmma_f32_16x16x32_f16(
                        false, A1.v, false, B.v, (short)0, acc2[t][1], false, false);
                }
            }
            // h0 = acc2 + b2s
#pragma unroll
            for (int t = 0; t < 4; ++t) {
                float bb = b2s[t * 16 + nl];
#pragma unroll
                for (int mb = 0; mb < 2; ++mb) {
#pragma unroll
                    for (int r = 0; r < 8; ++r) {
                        int e = e0 + r + 8 * hi + 16 * mb;
                        if (e < nEdges)
                            out[(size_t)e * 64 + t * 16 + nl] = acc2[t][mb][r] + bb;
                    }
                }
            }
        }

        // ---- layer 2, vector head: h_v(32x128) @ W2v(128x32), v0 epilogue ----
        {
            v8f accv[2][2];
#pragma unroll
            for (int t = 0; t < 2; ++t) { accv[t][0] = zero; accv[t][1] = zero; }
#pragma unroll
            for (int c = 0; c < 4; ++c) {
                AFrag A0, A1;
                const _Float16* hr0 = hT + nl * 256 + 128 + c * 32 + hi * 8;
                const _Float16* hr1 = hr0 + 16 * 256;
                A0.h[0] = *(const v8h*)hr0;  A0.h[1] = *(const v8h*)(hr0 + 16);
                A1.h[0] = *(const v8h*)hr1;  A1.h[1] = *(const v8h*)(hr1 + 16);
#pragma unroll
                for (int t = 0; t < 2; ++t) {
                    AFrag B;
                    const _Float16* br = w2f + (c * 6 + 4 + t) * 512 + lane * 16;
                    B.h[0] = *(const v8h*)br;
                    B.h[1] = *(const v8h*)(br + 8);
                    accv[t][0] = __builtin_amdgcn_wmma_f32_16x16x32_f16(
                        false, A0.v, false, B.v, (short)0, accv[t][0], false, false);
                    accv[t][1] = __builtin_amdgcn_wmma_f32_16x16x32_f16(
                        false, A1.v, false, B.v, (short)0, accv[t][1], false, false);
                }
            }
            // v0 = (accv + b2v) * (cos, sin)
#pragma unroll
            for (int t = 0; t < 2; ++t) {
                float bb = b2v[t * 16 + nl];
#pragma unroll
                for (int mb = 0; mb < 2; ++mb) {
#pragma unroll
                    for (int r = 0; r < 8; ++r) {
                        int m = r + 8 * hi + 16 * mb;
                        int e = e0 + m;
                        if (e < nEdges) {
                            float amp = accv[t][mb][r] + bb;
                            float2 cs = csL[m];
                            float2* po = (float2*)(out + V0_BASE + (size_t)e * 64 +
                                                   (size_t)(t * 16 + nl) * 2);
                            *po = make_float2(amp * cs.x, amp * cs.y);
                        }
                    }
                }
            }
        }
    }
}

extern "C" void kernel_launch(void* const* d_in, const int* in_sizes, int n_in,
                              void* d_out, int out_size, void* d_ws, size_t ws_size,
                              hipStream_t stream) {
    const long long* ep   = (const long long*)d_in[0];   // endpoints int64 [E,2]
    const float* lens     = (const float*)d_in[1];
    const float* theta    = (const float*)d_in[2];
    const float* feat     = (const float*)d_in[3];
    const float* W1s      = (const float*)d_in[4];
    const float* b1s      = (const float*)d_in[5];
    const float* W2s      = (const float*)d_in[6];
    const float* b2s      = (const float*)d_in[7];
    const float* W1v      = (const float*)d_in[8];
    const float* b1v      = (const float*)d_in[9];
    const float* W2v      = (const float*)d_in[10];
    const float* b2v      = (const float*)d_in[11];

    int nEdges = in_sizes[0] / 2;
    int nTiles = (nEdges + TILE_M - 1) / TILE_M;         // 25000
    int blocks = (nTiles + NWAVES - 1) / NWAVES;         // 6250

    prep_weights_kernel<<<64, 256, 0, stream>>>(W1s, b1s, W2s, b2s,
                                                W1v, b1v, W2v, b2v,
                                                (char*)d_ws);

    edge_mlp_kernel<<<blocks, NWAVES * 32, LDS_TOTAL, stream>>>(
        ep, lens, theta, feat, (const char*)d_ws, (float*)d_out,
        nEdges, nTiles);
}
